// CTRNN_47502338294089
// MI455X (gfx1250) — matrix operand order repaired and checked by reading, not compiled, not run on hardware
//
#include <hip/hip_runtime.h>
#include <hip/hip_bf16.h>

// CTRNN persistent fused kernel for MI455X (gfx1250, wave32, WMMA).
// SEQ=1024, BATCH=512, IN=128, HID=256, ALPHA=0.2
#define SEQ   1024
#define BATCH 512
#define IND   128
#define HID   256
#define BT    16          // batch tile per workgroup
#define NTHREADS 256      // 8 waves of 32

typedef __attribute__((ext_vector_type(16))) _Float16 v16h;
typedef __attribute__((ext_vector_type(8)))  float    v8f;

#if __has_builtin(__builtin_amdgcn_global_load_async_to_lds_b128) && \
    __has_builtin(__builtin_amdgcn_s_wait_asynccnt)
#define HAVE_ASYNC_LDS 1
// The builtin takes pointers to GCC-style int 4-vectors: AS1 (global) source,
// AS3 (LDS) destination (clang prints AS1 as `__device__` in diagnostics).
typedef int afi4 __attribute__((vector_size(16)));
typedef __attribute__((address_space(1))) afi4 ga_i4;
typedef __attribute__((address_space(3))) afi4 la_i4;
#else
#define HAVE_ASYNC_LDS 0
#endif

// A-matrix stored in plain K order within each 32-K block: lanes 0-15 read
// K 0..15, lanes 16-31 read K 16..31 (matches B fragments).
__device__ __forceinline__ int permA(int kl) {
  return kl;
}

// Load 16 contiguous halves (32 bytes) from LDS as a v16h fragment.
__device__ __forceinline__ v16h ld_frag(const _Float16* p) {
  v16h r;
  const uint4* q = (const uint4*)p;
  ((uint4*)&r)[0] = q[0];
  ((uint4*)&r)[1] = q[1];
  return r;
}

__device__ __forceinline__ v8f wmma_f16(v16h a, v16h b, v8f c) {
  return __builtin_amdgcn_wmma_f32_16x16x32_f16(
      /*neg_a=*/false, a, /*neg_b=*/false, b,
      /*c_mod=*/(short)0, c, /*reuse_a=*/false, /*reuse_b=*/false);
}

__global__ __launch_bounds__(NTHREADS, 1)
void ctrnn_persistent(const float* __restrict__ input,
                      const float* __restrict__ hidden,
                      const float* __restrict__ W_in,
                      const float* __restrict__ b_in,
                      const float* __restrict__ W_hh,
                      const float* __restrict__ b_hh,
                      float* __restrict__ out) {
  // LDS: W_hh f16 128KB + h 8KB + x(f16) 4KB + x(raw f32) 8KB + bias 1KB ~ 149KB
  __shared__ __align__(32) _Float16 sWhh[8][HID][32];  // [kb][n][klocal]
  __shared__ __align__(32) _Float16 sH[BT][8][32];     // [m][kb][permA pos]
  __shared__ __align__(32) _Float16 sX[BT][4][32];     // [m][kb][permA pos]
  __shared__ __align__(16) float    sXraw[BT * IND];   // raw f32 async staging
  __shared__ float sBias[HID];

  const int tid  = threadIdx.x;
  const int wave = tid >> 5;
  const int lane = tid & 31;
  const int hi   = lane >> 4;
  const int ln   = lane & 15;
  const int bt0  = blockIdx.x * BT;
  const int n0   = wave * 32 + ln;   // this lane's column in its wave's slice

  // ---- stage W_hh -> LDS f16, B-fragment layout (plain klocal order) ----
  for (int idx = tid; idx < HID * HID; idx += NTHREADS) {
    int k = idx >> 8;        // row (K)
    int n = idx & 255;       // col (N), coalesced over tid
    sWhh[k >> 5][n][k & 31] = (_Float16)W_hh[idx];
  }
  if (tid < HID) sBias[tid] = b_in[tid] + b_hh[tid];

  // ---- stage initial hidden state -> sH (A-fragment layout, f16) ----
  for (int idx = tid; idx < BT * HID; idx += NTHREADS) {
    int mrow = idx >> 8;
    int k    = idx & 255;
    float v  = hidden[(size_t)(bt0 + mrow) * HID + k];
    sH[mrow][k >> 5][permA(k & 31)] = (_Float16)v;
  }

  // ---- stage input tile for s=0 -> sX ----
  #pragma unroll
  for (int j = 0; j < 8; ++j) {
    int idx  = j * NTHREADS + tid;
    int mrow = idx >> 7;
    int k    = idx & 127;
    float v  = input[(size_t)(bt0 + mrow) * IND + k];
    sX[mrow][k >> 5][permA(k & 31)] = (_Float16)v;
  }

  // ---- W_in B-fragments resident in registers (reused all 1024 steps) ----
  // B-frag element e of VGPR-pair layout maps to K = kb*32 + hi*16 + e.
  v16h winf[4][2];
  for (int kb = 0; kb < 4; ++kb)
    for (int t = 0; t < 2; ++t)
      for (int e = 0; e < 16; ++e)
        winf[kb][t][e] =
            (_Float16)W_in[(size_t)(kb * 32 + hi * 16 + e) * HID + (n0 + t * 16)];

  // ---- hidden state in f32 C/D fragment layout: elem j -> row j+8*hi, col n0(+16)
  v8f hc0, hc1;
  #pragma unroll
  for (int j = 0; j < 8; ++j) {
    size_t row = (size_t)(bt0 + j + 8 * hi) * HID;
    hc0[j] = hidden[row + n0];
    hc1[j] = hidden[row + n0 + 16];
  }

  __syncthreads();

  const float bias0 = sBias[n0];
  const float bias1 = sBias[n0 + 16];

  for (int s = 0; s < SEQ; ++s) {
    // ---- software pipeline: fetch next timestep's input tile ----
#if HAVE_ASYNC_LDS
    // Async DMA global -> LDS (raw f32), tracked on ASYNCcnt; the 24 WMMAs
    // below cover the latency. Each thread owns floats [(j*256+tid)*4, +4).
    if (s + 1 < SEQ) {
      #pragma unroll
      for (int j = 0; j < 2; ++j) {
        int f = (j * NTHREADS + tid) * 4;   // flat float index in 16x128 tile
        int mrow = f >> 7;
        int k    = f & 127;
        const float* gp =
            input + ((size_t)(s + 1) * BATCH + bt0 + mrow) * IND + k;
        afi4* gpg = (afi4*)(uintptr_t)gp;       // generic, exact pointee type
        afi4* lpg = (afi4*)&sXraw[f];           // generic, exact pointee type
        __builtin_amdgcn_global_load_async_to_lds_b128(
            (ga_i4*)gpg, (la_i4*)lpg, 0, 0);
      }
    }
#else
    float xr[8];
    if (s + 1 < SEQ) {
      #pragma unroll
      for (int j = 0; j < 8; ++j) {
        int idx  = j * NTHREADS + tid;
        int mrow = idx >> 7;
        int k    = idx & 127;
        xr[j] = input[((size_t)(s + 1) * BATCH + bt0 + mrow) * IND + k];
      }
    }
#endif

    v8f acc0 = {};
    v8f acc1 = {};

    // Input projection: x[s] (16x128) @ W_in slice (128x32) — B from registers
    #pragma unroll
    for (int kb = 0; kb < 4; ++kb) {
      v16h a = ld_frag(&sX[ln][kb][hi * 16]);
      acc0 = wmma_f16(a, winf[kb][0], acc0);
      acc1 = wmma_f16(a, winf[kb][1], acc1);
    }
    // Recurrent projection: h (16x256) @ W_hh slice (256x32) — B from LDS
    #pragma unroll
    for (int kb = 0; kb < 8; ++kb) {
      v16h a  = ld_frag(&sH[ln][kb][hi * 16]);
      v16h b0 = ld_frag(&sWhh[kb][n0][hi * 16]);
      v16h b1 = ld_frag(&sWhh[kb][n0 + 16][hi * 16]);
      acc0 = wmma_f16(a, b0, acc0);
      acc1 = wmma_f16(a, b1, acc1);
    }

    // h_new = 0.8*h + 0.2*relu(pre + bias)
    #pragma unroll
    for (int j = 0; j < 8; ++j) {
      float p0 = fmaxf(acc0[j] + bias0, 0.0f);
      float p1 = fmaxf(acc1[j] + bias1, 0.0f);
      hc0[j] = 0.8f * hc0[j] + 0.2f * p0;
      hc1[j] = 0.8f * hc1[j] + 0.2f * p1;
    }

    // Stream output[s] (coalesced in n across lanes 0-15)
    float* outS = out + (size_t)s * BATCH * HID;
    #pragma unroll
    for (int j = 0; j < 8; ++j) {
      size_t row = (size_t)(bt0 + j + 8 * hi) * HID;
      outS[row + n0]      = hc0[j];
      outS[row + n0 + 16] = hc1[j];
    }

    __syncthreads();  // all A-fragment reads of sH/sX complete

    // Publish h_new as next step's A matrix (f16, plain K order).
    // kb = wave for both tiles; t=0 -> klocal ln, t=1 -> klocal ln+16.
    #pragma unroll
    for (int j = 0; j < 8; ++j) {
      int r  = j + 8 * hi;
      int k0 = n0;
      int k1 = n0 + 16;
      sH[r][k0 >> 5][permA(k0 & 31)] = (_Float16)hc0[j];
      sH[r][k1 >> 5][permA(k1 & 31)] = (_Float16)hc1[j];
    }

    // Commit next input tile into f16 fragment layout
    if (s + 1 < SEQ) {
#if HAVE_ASYNC_LDS
      __builtin_amdgcn_s_wait_asynccnt(0);  // own async b128s have landed
      #pragma unroll
      for (int j = 0; j < 2; ++j) {
        int f = (j * NTHREADS + tid) * 4;
        float4 v = *(const float4*)&sXraw[f];
        int mrow = f >> 7;
        int kbase = f & 127;
        sX[mrow][kbase >> 5][permA((kbase + 0) & 31)] = (_Float16)v.x;
        sX[mrow][kbase >> 5][permA((kbase + 1) & 31)] = (_Float16)v.y;
        sX[mrow][kbase >> 5][permA((kbase + 2) & 31)] = (_Float16)v.z;
        sX[mrow][kbase >> 5][permA((kbase + 3) & 31)] = (_Float16)v.w;
      }
#else
      #pragma unroll
      for (int j = 0; j < 8; ++j) {
        int idx  = j * NTHREADS + tid;
        int mrow = idx >> 7;
        int k    = idx & 127;
        sX[mrow][k >> 5][permA(k & 31)] = (_Float16)xr[j];
      }
#endif
    }
    __syncthreads();
  }

  // Final hidden state appended after output
  float* outH = out + (size_t)SEQ * BATCH * HID;
  #pragma unroll
  for (int j = 0; j < 8; ++j) {
    size_t row = (size_t)(bt0 + j + 8 * hi) * HID;
    outH[row + n0]      = hc0[j];
    outH[row + n0 + 16] = hc1[j];
  }
}

extern "C" void kernel_launch(void* const* d_in, const int* in_sizes, int n_in,
                              void* d_out, int out_size, void* d_ws, size_t ws_size,
                              hipStream_t stream) {
  (void)in_sizes; (void)n_in; (void)out_size; (void)d_ws; (void)ws_size;
  const float* input  = (const float*)d_in[0];
  const float* hidden = (const float*)d_in[1];
  const float* W_in   = (const float*)d_in[2];
  const float* b_in   = (const float*)d_in[3];
  const float* W_hh   = (const float*)d_in[4];
  const float* b_hh   = (const float*)d_in[5];
  float* out = (float*)d_out;

  dim3 grid(BATCH / BT);   // 32 persistent workgroups
  dim3 block(NTHREADS);    // 8 waves (wave32)
  ctrnn_persistent<<<grid, block, 0, stream>>>(input, hidden, W_in, b_in,
                                               W_hh, b_hh, out);
}
